// tmgp_sg_2680059593368
// MI455X (gfx1250) — matrix operand order repaired and checked by reading, compiled, not run on hardware
//
#include <hip/hip_runtime.h>

typedef __attribute__((ext_vector_type(16))) _Float16 v16h;
typedef __attribute__((ext_vector_type(8)))  _Float16 v8h;
typedef __attribute__((ext_vector_type(8)))  float    v8f;

#define N_ROWS 16384
#define DIMS   8
#define M_PTS  1024
#define ROWS_PER_BLOCK 64
#define SK_STRIDE 1032   // 1024 + 8 halves (16B) pad: conflict-free ds_load_b128 A reads

// ---------------- Kernel 1: per-feature min / max over the batch ----------------
__global__ __launch_bounds__(256) void minmax_kernel(const float* __restrict__ x,
                                                     float* __restrict__ mnmx) {
  __shared__ float smn[256][8];
  __shared__ float smx[256][8];
  float mn[8], mx[8];
#pragma unroll
  for (int d = 0; d < 8; ++d) { mn[d] = 3.402823466e38f; mx[d] = -3.402823466e38f; }
  for (int r = threadIdx.x; r < N_ROWS; r += 256) {
    const float4* p = (const float4*)(x + (size_t)r * 8);
    float4 a = p[0], b = p[1];
    float v[8] = {a.x, a.y, a.z, a.w, b.x, b.y, b.z, b.w};
#pragma unroll
    for (int d = 0; d < 8; ++d) { mn[d] = fminf(mn[d], v[d]); mx[d] = fmaxf(mx[d], v[d]); }
  }
#pragma unroll
  for (int d = 0; d < 8; ++d) { smn[threadIdx.x][d] = mn[d]; smx[threadIdx.x][d] = mx[d]; }
  __syncthreads();
  if (threadIdx.x < 8) {
    const int d = threadIdx.x;
    float a = smn[0][d], b = smx[0][d];
    for (int t = 1; t < 256; ++t) { a = fminf(a, smn[t][d]); b = fmaxf(b, smx[t][d]); }
    mnmx[d] = a;
    mnmx[8 + d] = b;
  }
}

// ---------------- Kernel 2: pack chol_inv (f32 row-major) into f16 B fragments ----------------
// Fragment f = kt*64 + nt covers B = chol_inv[kt*32 .. +32, nt*16 .. +16].
// Lane l (g = l>>4, n = l&15) holds 16 halves: K = g*16 + j (j = 0..15) at column n,
// stored contiguously (32 B per lane) so the GEMM does two global_load_b128 per fragment.
__global__ __launch_bounds__(256) void packB_kernel(const float* __restrict__ ci,
                                                    _Float16* __restrict__ Bp) {
  const int idx = blockIdx.x * 256 + threadIdx.x;   // 0 .. 131071
  const int l  = idx & 31;
  const int f  = idx >> 5;
  const int nt = f & 63;
  const int kt = f >> 6;
  const int g  = l >> 4;
  const int n  = l & 15;
  const float* src = ci + (size_t)(kt * 32 + g * 16) * M_PTS + nt * 16 + n;
  v16h vals;
#pragma unroll
  for (int j = 0; j < 16; ++j) vals[j] = (_Float16)src[(size_t)j * M_PTS];
  *(v16h*)(Bp + (size_t)idx * 16) = vals;
}

// ---------------- Kernel 3: fused k_star tile (LDS) + WMMA GEMM ----------------
__global__ __launch_bounds__(128) void fused_kernel(const float* __restrict__ x,
                                                    const float* __restrict__ dp,
                                                    const float* __restrict__ mnmx,
                                                    const _Float16* __restrict__ Bp,
                                                    float* __restrict__ out) {
  __shared__ _Float16 sk[ROWS_PER_BLOCK][SK_STRIDE];   // 132096 B
  __shared__ _Float16 sdp[M_PTS * DIMS];               //  16384 B

  const int tid = threadIdx.x;
  const int r0  = blockIdx.x * ROWS_PER_BLOCK;

  // stage design points into LDS as f16
  for (int i = tid; i < M_PTS * DIMS; i += 128) sdp[i] = (_Float16)dp[i];

  // per-feature affine rescale parameters (uniform -> scalar loads)
  float mn[8], sc[8];
#pragma unroll
  for (int d = 0; d < 8; ++d) { mn[d] = mnmx[d]; sc[d] = 1.0f / (mnmx[8 + d] - mnmx[d]); }

  __syncthreads();

  // Phase A: k_star tile [64 x 1024] -> LDS (f16). 2 threads per row, 512 m each.
  {
    const int rl   = tid >> 1;
    const int m0   = (tid & 1) * 512;
    const float4* px = (const float4*)(x + (size_t)(r0 + rl) * 8);
    float4 a = px[0], b = px[1];
    float xs[8] = {a.x, a.y, a.z, a.w, b.x, b.y, b.z, b.w};
#pragma unroll
    for (int d = 0; d < 8; ++d) xs[d] = (xs[d] - mn[d]) * sc[d];
    for (int m = m0; m < m0 + 512; ++m) {
      const _Float16* dpm = &sdp[m * 8];
      float l1 = 0.0f;
#pragma unroll
      for (int d = 0; d < 8; ++d) l1 += fabsf(xs[d] - (float)dpm[d]);
      sk[rl][m] = (_Float16)__expf(-l1);
    }
  }
  __syncthreads();

  // Phase B: GEMM. Wave w strides over N tiles; 4 row-subtiles per wave reuse each B frag.
  const int w  = tid >> 5;
  const int l  = tid & 31;
  const int g  = l >> 4;
  const int hl = l & 15;

  for (int i = 0; i < 16; ++i) {
    const int nt = w + 4 * i;
    v8f acc0 = {}, acc1 = {}, acc2 = {}, acc3 = {};
#pragma unroll 2
    for (int kt = 0; kt < 32; ++kt) {
      union { v16h v; v8h h[2]; } bf, af;
      const v8h* bp = (const v8h*)(Bp + ((size_t)(kt * 64 + nt) * 32 + l) * 16);
      bf.h[0] = bp[0];
      bf.h[1] = bp[1];
      // prefetch next B fragment (clamped so no tail peel, stays inside Bp)
      const int ktn = (kt < 31) ? (kt + 1) : 31;
      __builtin_prefetch(Bp + ((size_t)(ktn * 64 + nt) * 32 + l) * 16, 0, 0);
      const int c0 = kt * 32 + g * 16;  // A layout: lanes 0-15 K=0..15, 16-31 K=16..31
      af.h[0] = *(const v8h*)&sk[hl][c0];
      af.h[1] = *(const v8h*)&sk[hl][c0 + 8];
      acc0 = __builtin_amdgcn_wmma_f32_16x16x32_f16(false, af.v, false, bf.v, (short)0, acc0, false, false);
      af.h[0] = *(const v8h*)&sk[16 + hl][c0];
      af.h[1] = *(const v8h*)&sk[16 + hl][c0 + 8];
      acc1 = __builtin_amdgcn_wmma_f32_16x16x32_f16(false, af.v, false, bf.v, (short)0, acc1, false, false);
      af.h[0] = *(const v8h*)&sk[32 + hl][c0];
      af.h[1] = *(const v8h*)&sk[32 + hl][c0 + 8];
      acc2 = __builtin_amdgcn_wmma_f32_16x16x32_f16(false, af.v, false, bf.v, (short)0, acc2, false, false);
      af.h[0] = *(const v8h*)&sk[48 + hl][c0];
      af.h[1] = *(const v8h*)&sk[48 + hl][c0 + 8];
      acc3 = __builtin_amdgcn_wmma_f32_16x16x32_f16(false, af.v, false, bf.v, (short)0, acc3, false, false);
    }
    // C/D layout: VGPR i -> row (g*8 + i) of the 16x16 tile, column hl.
    // Output is write-once / never-reread: NT stores keep Bp + k_star sources in L2.
    float* op = out + (size_t)(r0 + g * 8) * M_PTS + nt * 16 + hl;
#pragma unroll
    for (int i2 = 0; i2 < 8; ++i2) {
      __builtin_nontemporal_store(acc0[i2], op + (size_t)(i2) * M_PTS);
      __builtin_nontemporal_store(acc1[i2], op + (size_t)(16 + i2) * M_PTS);
      __builtin_nontemporal_store(acc2[i2], op + (size_t)(32 + i2) * M_PTS);
      __builtin_nontemporal_store(acc3[i2], op + (size_t)(48 + i2) * M_PTS);
    }
  }
}

extern "C" void kernel_launch(void* const* d_in, const int* in_sizes, int n_in,
                              void* d_out, int out_size, void* d_ws, size_t ws_size,
                              hipStream_t stream) {
  const float* x  = (const float*)d_in[0];          // [16384, 8]
  const float* dp = (const float*)d_in[1];          // [1024, 8]
  const float* ci = (const float*)d_in[2];          // [1024, 1024]
  float* out = (float*)d_out;                       // [16384, 1024]

  float*    mnmx = (float*)d_ws;                            // 16 floats
  _Float16* Bp   = (_Float16*)((char*)d_ws + 256);          // 2 MB packed B

  minmax_kernel<<<1, 256, 0, stream>>>(x, mnmx);
  packB_kernel<<<512, 256, 0, stream>>>(ci, Bp);
  fused_kernel<<<N_ROWS / ROWS_PER_BLOCK, 128, 0, stream>>>(x, dp, mnmx, Bp, out);
}
